// AttnBlock_3779571220706
// MI455X (gfx1250) — compile-verified, matrix operand order checked by reading, not run on hardware
//
#include <hip/hip_runtime.h>
#include <math.h>

typedef __attribute__((ext_vector_type(16))) __bf16 v16bf;
typedef __attribute__((ext_vector_type(8)))  __bf16 v8bf;
typedef __attribute__((ext_vector_type(8)))  float  v8f;

#define BATCH 4
#define CDIM  512
#define HWN   4096

__device__ inline v16bf cat16(v8bf lo, v8bf hi) {
    return __builtin_shufflevector(lo, hi, 0,1,2,3,4,5,6,7,8,9,10,11,12,13,14,15);
}

// Async DMA: 16 bytes per lane, global -> LDS, tracked by ASYNCcnt.
__device__ inline void async_copy_b128(void* lds_ptr, const void* gptr) {
    unsigned lds_off = (unsigned)(size_t)lds_ptr;
    asm volatile("global_load_async_to_lds_b128 %0, %1, off"
                 :: "v"(lds_off), "v"(gptr) : "memory");
}
__device__ inline void wait_async_le4() {
    asm volatile("s_wait_asynccnt 4" ::: "memory");
}
__device__ inline void wait_async_le8() {
    asm volatile("s_wait_asynccnt 8" ::: "memory");
}

// 16x32 bf16 fragment load (A-layout; also used for B staged as [n][k]).
// lanes 0-15: row=lane, K chunks {0..7},{16..23}; lanes 16-31: row=lane-16, {8..15},{24..31}.
__device__ inline v16bf load_frag(const __bf16* base, int stride, int lane) {
    int r = lane & 15;
    int off = (lane < 16) ? 0 : 8;
    const __bf16* p = base + (size_t)r * stride + off;
    v8bf lo = *(const v8bf*)(p);
    v8bf hi = *(const v8bf*)(p + 16);
    return cat16(lo, hi);
}

__device__ inline v8f wmma_bf16(v16bf a, v16bf b, v8f c) {
    return __builtin_amdgcn_wmma_f32_16x16x32_bf16(false, a, false, b, (short)0, c, false, false);
}

// ---------------- GroupNorm: x[b,c,h,w] -> hnT[b, hw, c] (bf16, transposed) --------------
__global__ void __launch_bounds__(256)
groupnorm_kernel(const float* __restrict__ x, const float* __restrict__ gamma,
                 const float* __restrict__ beta, __bf16* __restrict__ hnT) {
    __shared__ float red[256], red2[256];
    __shared__ float s_mean, s_rstd;
    int b = blockIdx.x >> 5;
    int grp = blockIdx.x & 31;
    const float* xb = x + ((size_t)b * CDIM + grp * 16) * HWN;
    float s = 0.f, s2 = 0.f;
    for (int i = threadIdx.x; i < 16 * HWN; i += 256) {
        float v = xb[i];
        s += v; s2 += v * v;
    }
    red[threadIdx.x] = s; red2[threadIdx.x] = s2;
    __syncthreads();
    for (int off = 128; off > 0; off >>= 1) {
        if ((int)threadIdx.x < off) {
            red[threadIdx.x]  += red[threadIdx.x + off];
            red2[threadIdx.x] += red2[threadIdx.x + off];
        }
        __syncthreads();
    }
    if (threadIdx.x == 0) {
        float inv_n = 1.f / (16.f * HWN);
        float mean = red[0] * inv_n;
        float var = red2[0] * inv_n - mean * mean;
        s_mean = mean;
        s_rstd = rsqrtf(var + 1e-6f);
    }
    __syncthreads();
    float mean = s_mean, rstd = s_rstd;
    for (int i = threadIdx.x; i < 16 * HWN; i += 256) {
        int cl = i >> 12;
        int c = grp * 16 + cl;
        int pix = i & (HWN - 1);
        float v = (xb[i] - mean) * rstd * gamma[c] + beta[c];
        hnT[((size_t)b * HWN + pix) * CDIM + c] = (__bf16)v;
    }
}

// ---------------- fp32 -> bf16 weight conversion (wq|wk|wv|wp) ----------------
__global__ void __launch_bounds__(256)
wcvt_kernel(const float* __restrict__ w0, const float* __restrict__ w1,
            const float* __restrict__ w2, const float* __restrict__ w3,
            __bf16* __restrict__ o0, __bf16* __restrict__ o1,
            __bf16* __restrict__ o2, __bf16* __restrict__ o3) {
    int id = blockIdx.x * 256 + threadIdx.x;      // 0 .. 4*512*512-1
    int mat = id >> 18, off = id & 262143;
    const float* s = (mat == 0) ? w0 : (mat == 1) ? w1 : (mat == 2) ? w2 : w3;
    __bf16*      o = (mat == 0) ? o0 : (mat == 1) ? o1 : (mat == 2) ? o2 : o3;
    o[off] = (__bf16)s[off];
}

// ---------------- GEMM: out[b,m,n] = scale*(sum_k Wb[m,k]*XT[b,n,k] + bias[m]) (+res) ----
// Wb bf16 [512,512]; XT bf16 [b,4096,512]. Double-buffered async-to-LDS staging.
template <bool TRANS_OUT, bool BF16OUT, bool RES>
__global__ void __launch_bounds__(256)
gemm_kernel(const __bf16* __restrict__ Wb, const __bf16* __restrict__ XTall,
            const float* __restrict__ bias, void* __restrict__ outv,
            const float* __restrict__ resAll, float outScale) {
    constexpr int M = 512, K = 512, N = 4096;
    __shared__ __attribute__((aligned(16))) __bf16 As[2][128 * 32]; // [m][k]
    __shared__ __attribute__((aligned(16))) __bf16 Bs[2][128 * 32]; // [n][k]
    int b = blockIdx.z;
    const __bf16* XT = XTall + (size_t)b * N * K;
    int m0 = blockIdx.y * 128, n0 = blockIdx.x * 128;
    int tid = threadIdx.x, lane = tid & 31, w = tid >> 5;
    int wm = w >> 2, wn = w & 3;                 // wave tile: 64 rows x 32 cols
    int r0 = tid >> 2, seg = (tid & 3) * 8;      // staging: rows r0, r0+64; 16B chunks

    auto stage = [&](int k0, int buf) {
        async_copy_b128(&As[buf][r0 * 32 + seg],        Wb + (size_t)(m0 + r0) * K + k0 + seg);
        async_copy_b128(&As[buf][(r0 + 64) * 32 + seg], Wb + (size_t)(m0 + r0 + 64) * K + k0 + seg);
        async_copy_b128(&Bs[buf][r0 * 32 + seg],        XT + (size_t)(n0 + r0) * K + k0 + seg);
        async_copy_b128(&Bs[buf][(r0 + 64) * 32 + seg], XT + (size_t)(n0 + r0 + 64) * K + k0 + seg);
    };

    v8f acc[4][2] = {};
    stage(0, 0);                                  // prologue: tile 0
    for (int kt = 0; kt < 16; ++kt) {
        int buf = kt & 1;
        stage(((kt + 1) & 15) * 32, buf ^ 1);     // prefetch next (tail wraps, harmless)
        wait_async_le4();                         // oldest 4 (current tile) complete
        __syncthreads();
        v16bf aF[4];
#pragma unroll
        for (int mi = 0; mi < 4; mi++)
            aF[mi] = load_frag(&As[buf][(64 * wm + 16 * mi) * 32], 32, lane);
#pragma unroll
        for (int ni = 0; ni < 2; ni++) {
            v16bf bF = load_frag(&Bs[buf][(32 * wn + 16 * ni) * 32], 32, lane);
#pragma unroll
            for (int mi = 0; mi < 4; mi++)
                acc[mi][ni] = wmma_bf16(aF[mi], bF, acc[mi][ni]);
        }
        __syncthreads();                          // protects ping-pong write-after-read
    }
    int rsel = (lane >= 16) ? 8 : 0, nl = lane & 15;
#pragma unroll
    for (int mi = 0; mi < 4; mi++)
#pragma unroll
        for (int ni = 0; ni < 2; ni++)
#pragma unroll
            for (int r = 0; r < 8; r++) {
                int m = m0 + 64 * wm + 16 * mi + r + rsel;
                int n = n0 + 32 * wn + 16 * ni + nl;
                float val = (acc[mi][ni][r] + bias[m]) * outScale;
                size_t oidx = TRANS_OUT ? ((size_t)b * N * M + (size_t)n * M + m)
                                        : ((size_t)b * M * N + (size_t)m * N + n);
                if constexpr (RES) val += resAll[oidx];
                if constexpr (BF16OUT) ((__bf16*)outv)[oidx] = (__bf16)val;
                else                   ((float*)outv)[oidx] = val;
            }
}

// ---------------- Fused flash attention ----------------
// qT/kT bf16 [b, HW, C] (q pre-scaled by 1/sqrt(C)), v bf16 [b, C, HW].
// Output oT bf16 [b, HW, C]. Block = 512 thr = 16 waves; 64 query rows per block.
// K tiles double-buffered via async DMA.
__global__ void __launch_bounds__(512)
attn_kernel(const __bf16* __restrict__ qTa, const __bf16* __restrict__ kTa,
            const __bf16* __restrict__ va, __bf16* __restrict__ oTa) {
    constexpr int QS = 520;  // 512 + 8 pad
    constexpr int SS = 65;
    constexpr int PS = 72;
    __shared__ __attribute__((aligned(16))) __bf16 Qs[64 * QS];     // [i][c]
    __shared__ __attribute__((aligned(16))) __bf16 Ks[2][64 * QS];  // [j][c], ping-pong
    __shared__ float Sss[64 * SS];
    __shared__ __attribute__((aligned(16))) __bf16 Psh[64 * PS];
    __shared__ float m_run[64], l_run[64], alpha_s[64];

    int b = blockIdx.y;
    int i0 = blockIdx.x * 64;
    const __bf16* qT = qTa + (size_t)b * HWN * CDIM;
    const __bf16* kT = kTa + (size_t)b * HWN * CDIM;
    const __bf16* v  = va  + (size_t)b * CDIM * HWN;
    int tid = threadIdx.x, lane = tid & 31, w = tid >> 5;
    int si = w >> 2, sj = w & 3;
    int rsel = (lane >= 16) ? 8 : 0, nl = lane & 15;

    auto stage_k = [&](int jtile, int buf) {      // 8 x 16B per thread
#pragma unroll
        for (int it = 0; it < 8; ++it) {
            int e = (it * 512 + tid) * 8;
            int row = e >> 9, col = e & 511;
            async_copy_b128(&Ks[buf][row * QS + col],
                            kT + (size_t)(jtile * 64 + row) * CDIM + col);
        }
    };

    // prologue: stage Q tile, then K tile 0 (in-order completion matters below)
#pragma unroll
    for (int it = 0; it < 8; ++it) {
        int e = (it * 512 + tid) * 8;
        int row = e >> 9, col = e & 511;
        async_copy_b128(Qs + row * QS + col, qT + (size_t)(i0 + row) * CDIM + col);
    }
    stage_k(0, 0);
    if (tid < 64) { m_run[tid] = -1e30f; l_run[tid] = 0.f; }
    v8f O[8] = {};

    for (int jt = 0; jt < 64; ++jt) {
        int j0 = jt * 64;
        int buf = jt & 1;
        stage_k((jt + 1) & 63, buf ^ 1);          // prefetch next (tail wraps, harmless)
        wait_async_le8();                         // Q + current K tile complete (in-order)
        __syncthreads();
        // S tile (16x16 per wave), K-loop over C
        v8f sAcc = {};
        for (int kc = 0; kc < CDIM; kc += 32) {
            v16bf aF = load_frag(Qs + (size_t)(16 * si) * QS + kc, QS, lane);
            v16bf bF = load_frag(&Ks[buf][(size_t)(16 * sj) * QS + kc], QS, lane);
            sAcc = wmma_bf16(aF, bF, sAcc);
        }
#pragma unroll
        for (int r = 0; r < 8; r++)
            Sss[(16 * si + r + rsel) * SS + 16 * sj + nl] = sAcc[r];
        __syncthreads();
        // online softmax: one thread per query row
        if (tid < 64) {
            int i = tid;
            float mold = m_run[i];
            float tmax = mold;
            for (int j = 0; j < 64; j++) tmax = fmaxf(tmax, Sss[i * SS + j]);
            float a = __expf(mold - tmax);
            float psum = 0.f;
            for (int j = 0; j < 64; j++) {
                float p = __expf(Sss[i * SS + j] - tmax);
                Psh[i * PS + j] = (__bf16)p;
                psum += p;
            }
            l_run[i] = a * l_run[i] + psum;
            m_run[i] = tmax;
            alpha_s[i] = a;
        }
        __syncthreads();
        // rescale O, accumulate P·V (V B-fragments straight from global rows)
        {
            float al[8];
#pragma unroll
            for (int r = 0; r < 8; r++) al[r] = alpha_s[16 * si + r + rsel];
#pragma unroll
            for (int ci = 0; ci < 8; ci++)
#pragma unroll
                for (int r = 0; r < 8; r++) O[ci][r] *= al[r];
#pragma unroll
            for (int ks = 0; ks < 2; ++ks) {
                v16bf aP = load_frag(Psh + (size_t)(16 * si) * PS + 32 * ks, PS, lane);
#pragma unroll
                for (int ci = 0; ci < 8; ci++) {
                    int c = 128 * sj + 16 * ci + nl;
                    const __bf16* vp = v + (size_t)c * HWN + j0 + 32 * ks + ((lane < 16) ? 0 : 8);
                    v8bf lo = *(const v8bf*)(vp);
                    v8bf hi = *(const v8bf*)(vp + 16);
                    O[ci] = wmma_bf16(aP, cat16(lo, hi), O[ci]);
                }
            }
        }
        __syncthreads();                           // protects K ping-pong + Psh reuse
    }
    // normalize + store transposed: oT[b][i][c]
#pragma unroll
    for (int ci = 0; ci < 8; ci++) {
        int c = 128 * sj + 16 * ci + nl;
#pragma unroll
        for (int r = 0; r < 8; r++) {
            int il = 16 * si + r + rsel;
            float val = O[ci][r] / l_run[il];
            oTa[(size_t)b * HWN * CDIM + (size_t)(i0 + il) * CDIM + c] = (__bf16)val;
        }
    }
}

extern "C" void kernel_launch(void* const* d_in, const int* in_sizes, int n_in,
                              void* d_out, int out_size, void* d_ws, size_t ws_size,
                              hipStream_t stream) {
    const float* x     = (const float*)d_in[0];
    const float* gamma = (const float*)d_in[1];
    const float* beta  = (const float*)d_in[2];
    const float* wq    = (const float*)d_in[3];
    const float* bq    = (const float*)d_in[4];
    const float* wk    = (const float*)d_in[5];
    const float* bk    = (const float*)d_in[6];
    const float* wv    = (const float*)d_in[7];
    const float* bv    = (const float*)d_in[8];
    const float* wp    = (const float*)d_in[9];
    const float* bp    = (const float*)d_in[10];
    float* out = (float*)d_out;

    char* ws = (char*)d_ws;
    const size_t SEG  = (size_t)BATCH * CDIM * HWN * sizeof(__bf16); // 16 MB
    const size_t WSEG = (size_t)CDIM * CDIM * sizeof(__bf16);        // 512 KB
    __bf16* hnT = (__bf16*)(ws);                 // [b][hw][c]
    __bf16* qT  = (__bf16*)(ws + SEG);           // [b][hw][c]
    __bf16* kT  = (__bf16*)(ws + 2 * SEG);       // [b][hw][c]
    __bf16* vb  = (__bf16*)(ws + 3 * SEG);       // [b][c][hw]
    __bf16* aT  = (__bf16*)(ws + 4 * SEG);       // [b][hw][c]
    __bf16* wqb = (__bf16*)(ws + 5 * SEG);
    __bf16* wkb = (__bf16*)(ws + 5 * SEG + WSEG);
    __bf16* wvb = (__bf16*)(ws + 5 * SEG + 2 * WSEG);
    __bf16* wpb = (__bf16*)(ws + 5 * SEG + 3 * WSEG);

    groupnorm_kernel<<<dim3(BATCH * 32), 256, 0, stream>>>(x, gamma, beta, hnT);
    wcvt_kernel<<<dim3(4 * 512 * 512 / 256), 256, 0, stream>>>(wq, wk, wv, wp, wqb, wkb, wvb, wpb);

    dim3 ggrid(HWN / 128, CDIM / 128, BATCH);
    const float qscale = 0.04419417382415922f; // 1/sqrt(512)
    gemm_kernel<true,  true,  false><<<ggrid, 256, 0, stream>>>(wqb, hnT, bq, (void*)qT, nullptr, qscale);
    gemm_kernel<true,  true,  false><<<ggrid, 256, 0, stream>>>(wkb, hnT, bk, (void*)kT, nullptr, 1.0f);
    gemm_kernel<false, true,  false><<<ggrid, 256, 0, stream>>>(wvb, hnT, bv, (void*)vb, nullptr, 1.0f);

    attn_kernel<<<dim3(HWN / 64, BATCH), 512, 0, stream>>>(qT, kT, vb, aT);

    gemm_kernel<false, false, true><<<ggrid, 256, 0, stream>>>(wpb, aT, bp, (void*)out, x, 1.0f);
}
